// FlashPhiAttention_72421738545646
// MI455X (gfx1250) — compile-verified, matrix operand order checked by reading, not run on hardware
//
#include <hip/hip_runtime.h>
#include <hip/hip_bf16.h>

// ---------------- problem constants ----------------
#define T_TOK   4096          // B*S
#define HIDDEN  2560
#define HEADS   32
#define HEAD_DIM 80
#define HDP     96            // head dim padded to multiple of 32
#define ROT     40
#define ROT_HALF 20
#define BATCH   2
#define SEQ     2048
#define QKV_W   (3*HIDDEN)    // 7680

typedef __attribute__((ext_vector_type(16))) __bf16 bf16x16;
typedef __attribute__((ext_vector_type(8)))  float  f32x8;

struct BF16Vec {
    union { uint4 u[2]; bf16x16 v; };
};

__device__ __forceinline__ unsigned short f2bf(float f) {
    union { float f; unsigned int u; } x; x.f = f;
    unsigned int r = x.u + 0x7FFFu + ((x.u >> 16) & 1u);   // round-to-nearest-even
    return (unsigned short)(r >> 16);
}
__device__ __forceinline__ float bf2f(unsigned short h) {
    union { unsigned int u; float f; } x; x.u = ((unsigned int)h) << 16;
    return x.f;
}

// ---------------- 1) f32 -> bf16 conversion ----------------
__global__ __launch_bounds__(256) void cvt_f32_bf16(const float* __restrict__ src,
                                                    unsigned short* __restrict__ dst,
                                                    int n) {
    int i = blockIdx.x * 256 + threadIdx.x;
    if (i < n) dst[i] = f2bf(src[i]);
}

// ---------------- 2/5) register-blocked GEMM ----------------
// C[M,N] = A[M,K] * B[N,K]^T + bias.  A,B bf16 row-major (K contiguous).
// Block = 256 threads = 8 waves arranged 2x4; macro tile 64(M) x 256(N).
// Each wave: 32x64 = 2x4 subtiles of 16x16 -> 8 WMMAs per 32-deep k-step
// from 2 A-operand + 4 B-operand loads (4x/2x register reuse).
template <bool OUT_F32>
__global__ __launch_bounds__(256) void gemm_wmma_bf16(const unsigned short* __restrict__ A,
                                                      const unsigned short* __restrict__ B,
                                                      const float* __restrict__ bias,
                                                      void* __restrict__ Cout,
                                                      int M, int N, int K) {
    const int lane = threadIdx.x & 31;
    const int w    = threadIdx.x >> 5;
    const int wr   = w >> 2;                   // 0..1  (M direction)
    const int wc   = w & 3;                    // 0..3  (N direction)
    const int mb   = blockIdx.y * 64 + wr * 32;
    const int nb   = blockIdx.x * 256 + wc * 64;
    const int lm   = lane & 15;
    const int koff = (lane >> 4) * 8;          // K sub-chunk per ISA A/B layout

    const unsigned short* Ap0 = A + (size_t)(mb + lm) * K + koff;
    const unsigned short* Ap1 = Ap0 + (size_t)16 * K;
    const unsigned short* Bp0 = B + (size_t)(nb + lm) * K + koff;
    const unsigned short* Bp1 = Bp0 + (size_t)16 * K;
    const unsigned short* Bp2 = Bp0 + (size_t)32 * K;
    const unsigned short* Bp3 = Bp0 + (size_t)48 * K;

    f32x8 acc[2][4];
    #pragma unroll
    for (int mi = 0; mi < 2; ++mi)
        #pragma unroll
        for (int ni = 0; ni < 4; ++ni) acc[mi][ni] = (f32x8){};

    for (int kt = 0; kt < K; kt += 32) {
        BF16Vec a0, a1, b0, b1, b2, b3;
        a0.u[0] = *(const uint4*)(Ap0);  a0.u[1] = *(const uint4*)(Ap0 + 16);
        a1.u[0] = *(const uint4*)(Ap1);  a1.u[1] = *(const uint4*)(Ap1 + 16);
        b0.u[0] = *(const uint4*)(Bp0);  b0.u[1] = *(const uint4*)(Bp0 + 16);
        b1.u[0] = *(const uint4*)(Bp1);  b1.u[1] = *(const uint4*)(Bp1 + 16);
        b2.u[0] = *(const uint4*)(Bp2);  b2.u[1] = *(const uint4*)(Bp2 + 16);
        b3.u[0] = *(const uint4*)(Bp3);  b3.u[1] = *(const uint4*)(Bp3 + 16);
        __builtin_prefetch(Ap0 + 128, 0, 0);   // global_prefetch_b8
        __builtin_prefetch(Bp0 + 128, 0, 0);

        acc[0][0] = __builtin_amdgcn_wmma_f32_16x16x32_bf16(false, a0.v, false, b0.v, (short)0, acc[0][0], false, false);
        acc[0][1] = __builtin_amdgcn_wmma_f32_16x16x32_bf16(false, a0.v, false, b1.v, (short)0, acc[0][1], false, false);
        acc[0][2] = __builtin_amdgcn_wmma_f32_16x16x32_bf16(false, a0.v, false, b2.v, (short)0, acc[0][2], false, false);
        acc[0][3] = __builtin_amdgcn_wmma_f32_16x16x32_bf16(false, a0.v, false, b3.v, (short)0, acc[0][3], false, false);
        acc[1][0] = __builtin_amdgcn_wmma_f32_16x16x32_bf16(false, a1.v, false, b0.v, (short)0, acc[1][0], false, false);
        acc[1][1] = __builtin_amdgcn_wmma_f32_16x16x32_bf16(false, a1.v, false, b1.v, (short)0, acc[1][1], false, false);
        acc[1][2] = __builtin_amdgcn_wmma_f32_16x16x32_bf16(false, a1.v, false, b2.v, (short)0, acc[1][2], false, false);
        acc[1][3] = __builtin_amdgcn_wmma_f32_16x16x32_bf16(false, a1.v, false, b3.v, (short)0, acc[1][3], false, false);

        Ap0 += 32; Ap1 += 32; Bp0 += 32; Bp1 += 32; Bp2 += 32; Bp3 += 32;
    }

    // C layout: lane holds col = n&15, rows = (lane>>4)*8 + i
    #pragma unroll
    for (int mi = 0; mi < 2; ++mi) {
        const int rbase = mb + mi * 16 + (lane >> 4) * 8;
        #pragma unroll
        for (int ni = 0; ni < 4; ++ni) {
            const int col = nb + ni * 16 + lm;
            const float bv = bias[col];
            if (OUT_F32) {
                float* C = (float*)Cout;
                #pragma unroll
                for (int i = 0; i < 8; ++i)
                    C[(size_t)(rbase + i) * N + col] = acc[mi][ni][i] + bv;
            } else {
                unsigned short* C = (unsigned short*)Cout;
                #pragma unroll
                for (int i = 0; i < 8; ++i)
                    C[(size_t)(rbase + i) * N + col] = f2bf(acc[mi][ni][i] + bv);
            }
        }
    }
}

// ---------------- 3) RoPE on Q and K (in-place on bf16 qkv) ----------------
__global__ __launch_bounds__(256) void rope_kernel(unsigned short* __restrict__ qkv,
                                                   const float* __restrict__ cosp,
                                                   const float* __restrict__ sinp) {
    int idx = blockIdx.x * 256 + threadIdx.x;
    const int total = T_TOK * HEADS * ROT_HALF;
    if (idx >= total) return;
    int j = idx % ROT_HALF;
    int r = idx / ROT_HALF;
    int h = r % HEADS;
    int t = r / HEADS;
    float c = cosp[t * ROT_HALF + j];
    float s = sinp[t * ROT_HALF + j];
    size_t base = (size_t)t * QKV_W + h * HEAD_DIM;
    {   // Q
        float x1 = bf2f(qkv[base + j]);
        float x2 = bf2f(qkv[base + ROT_HALF + j]);
        qkv[base + j]            = f2bf(x1 * c - x2 * s);
        qkv[base + ROT_HALF + j] = f2bf(x1 * s + x2 * c);
    }
    {   // K (offset HIDDEN columns)
        float x1 = bf2f(qkv[base + HIDDEN + j]);
        float x2 = bf2f(qkv[base + HIDDEN + ROT_HALF + j]);
        qkv[base + HIDDEN + j]            = f2bf(x1 * c - x2 * s);
        qkv[base + HIDDEN + ROT_HALF + j] = f2bf(x1 * s + x2 * c);
    }
}

// ---------------- 4) causal flash attention ----------------
// grid = (SEQ/128, HEADS, BATCH), block = 256 (8 waves).
// Wave w owns Q rows [qblk*128 + w*16, +16). K/V staged to LDS per 32-key step.
__global__ __launch_bounds__(256) void flash_attn(const unsigned short* __restrict__ qkv,
                                                  unsigned short* __restrict__ attn) {
    const int tid  = threadIdx.x;
    const int lane = tid & 31;
    const int w    = tid >> 5;
    const int head = blockIdx.y;
    const int bat  = blockIdx.z;
    const int qw   = blockIdx.x * 128 + w * 16;     // wave's first q row (in-batch)
    const int koff = (lane >> 4) * 8;
    const int n    = lane & 15;
    const int rb   = (lane >> 4) * 8;

    __shared__ unsigned short Ksh[32 * HDP];        // [key][d] padded
    __shared__ unsigned short Vsh[HDP * 32];        // [d][key] transposed
    __shared__ unsigned short Psh[8][16 * 32];      // per-wave P tile (A-layout staging)

    // ---- Q tile into registers (3 K-steps of 32, D padded to 96) ----
    BF16Vec qv[3];
    {
        const unsigned short* qp =
            qkv + (size_t)(bat * SEQ + qw + (lane & 15)) * QKV_W + head * HEAD_DIM;
        #pragma unroll
        for (int s = 0; s < 3; ++s) {
            int d0 = s * 32 + koff;
            qv[s].u[0] = (d0      < HEAD_DIM) ? *(const uint4*)(qp + d0)      : make_uint4(0,0,0,0);
            qv[s].u[1] = (d0 + 16 < HEAD_DIM) ? *(const uint4*)(qp + d0 + 16) : make_uint4(0,0,0,0);
        }
    }

    // ---- pad columns/rows (d in [80,96)) are loop-invariant: zero them once ----
    for (int e = tid; e < 512; e += 256) {          // Ksh: 32 keys x 16 pad cols
        int key = e >> 4, j = e & 15;
        Ksh[key * HDP + HEAD_DIM + j] = 0;
    }
    for (int e = tid; e < 512; e += 256)            // Vsh: rows 80..95 are contiguous
        Vsh[HEAD_DIM * 32 + e] = 0;

    float mrow[8], lrow[8];
    f32x8 acc[5];
    #pragma unroll
    for (int i = 0; i < 8; ++i) { mrow[i] = -1e30f; lrow[i] = 0.0f; }
    #pragma unroll
    for (int t = 0; t < 5; ++t) acc[t] = (f32x8){};

    const float scale = 0.11180339887498949f;       // 1/sqrt(80)
    const unsigned short* kbase = qkv + (size_t)(bat * SEQ) * QKV_W + HIDDEN     + head * HEAD_DIM;
    const unsigned short* vbase = qkv + (size_t)(bat * SEQ) * QKV_W + 2 * HIDDEN + head * HEAD_DIM;
    const int kend = blockIdx.x * 128 + 128;        // causal: last key needed by this block

    for (int kb = 0; kb < kend; kb += 32) {
        // ---- cooperative staging: K rows copied as 16B chunks ----
        for (int e = tid; e < 320; e += 256) {      // 32 keys x 10 chunks
            int key = e / 10, dv = e % 10;
            *(uint4*)(Ksh + key * HDP + dv * 8) =
                *(const uint4*)(kbase + (size_t)(kb + key) * QKV_W + dv * 8);
        }
        // ---- V: 16B coalesced load + 8-way transpose scatter into LDS ----
        for (int e = tid; e < 320; e += 256) {
            int key = e & 31, dv = e >> 5;
            uint4 val = *(const uint4*)(vbase + (size_t)(kb + key) * QKV_W + dv * 8);
            const unsigned short* vs = (const unsigned short*)&val;
            #pragma unroll
            for (int j = 0; j < 8; ++j)
                Vsh[(dv * 8 + j) * 32 + key] = vs[j];
        }
        __syncthreads();

        if (kb <= qw + 15) {                        // wave has unmasked keys in this step
            // ---- scores: S = Q * K^T (two 16-key n-tiles) ----
            f32x8 s0 = {}, s1 = {};
            #pragma unroll
            for (int s = 0; s < 3; ++s) {
                int d0 = s * 32 + koff;
                const unsigned short* kr0 = Ksh + n * HDP + d0;
                const unsigned short* kr1 = Ksh + (n + 16) * HDP + d0;
                BF16Vec b0, b1;
                b0.u[0] = *(const uint4*)(kr0); b0.u[1] = *(const uint4*)(kr0 + 16);
                b1.u[0] = *(const uint4*)(kr1); b1.u[1] = *(const uint4*)(kr1 + 16);
                s0 = __builtin_amdgcn_wmma_f32_16x16x32_bf16(false, qv[s].v, false, b0.v,
                                                             (short)0, s0, false, false);
                s1 = __builtin_amdgcn_wmma_f32_16x16x32_bf16(false, qv[s].v, false, b1.v,
                                                             (short)0, s1, false, false);
            }
            // ---- online softmax (rows live across 16-lane halves) ----
            #pragma unroll
            for (int i = 0; i < 8; ++i) {
                int qg = qw + rb + i;
                float x0 = s0[i] * scale; if (kb + n      > qg) x0 = -1e30f;
                float x1 = s1[i] * scale; if (kb + 16 + n > qg) x1 = -1e30f;
                float rmax = fmaxf(x0, x1);
                #pragma unroll
                for (int off = 1; off < 16; off <<= 1)
                    rmax = fmaxf(rmax, __shfl_xor(rmax, off));
                float mnew  = fmaxf(mrow[i], rmax);
                float alpha = __expf(mrow[i] - mnew);
                float p0 = __expf(x0 - mnew);
                float p1 = __expf(x1 - mnew);
                float ps = p0 + p1;
                #pragma unroll
                for (int off = 1; off < 16; off <<= 1)
                    ps += __shfl_xor(ps, off);
                lrow[i] = lrow[i] * alpha + ps;
                mrow[i] = mnew;
                #pragma unroll
                for (int t = 0; t < 5; ++t) acc[t][i] *= alpha;
                Psh[w][(rb + i) * 32 + n]      = f2bf(p0);
                Psh[w][(rb + i) * 32 + 16 + n] = f2bf(p1);
            }
            // ---- O += P * V  (P remapped C-layout -> A-layout via LDS) ----
            BF16Vec pa;
            const unsigned short* pp = Psh[w] + (lane & 15) * 32 + koff;
            pa.u[0] = *(const uint4*)(pp);
            pa.u[1] = *(const uint4*)(pp + 16);
            #pragma unroll
            for (int nt = 0; nt < 5; ++nt) {
                const unsigned short* vp = Vsh + (nt * 16 + n) * 32 + koff;  // transposed V
                BF16Vec b;
                b.u[0] = *(const uint4*)(vp);
                b.u[1] = *(const uint4*)(vp + 16);
                acc[nt] = __builtin_amdgcn_wmma_f32_16x16x32_bf16(false, pa.v, false, b.v,
                                                                  (short)0, acc[nt], false, false);
            }
        }
        __syncthreads();
    }

    // ---- epilogue: O /= l, write bf16 [t, h*80+d] ----
    #pragma unroll
    for (int nt = 0; nt < 5; ++nt) {
        int d = nt * 16 + n;
        #pragma unroll
        for (int i = 0; i < 8; ++i) {
            int row = qw + rb + i;
            attn[(size_t)(bat * SEQ + row) * HIDDEN + head * HEAD_DIM + d] =
                f2bf(acc[nt][i] / lrow[i]);
        }
    }
}

// ---------------- host-side launcher ----------------
extern "C" void kernel_launch(void* const* d_in, const int* in_sizes, int n_in,
                              void* d_out, int out_size, void* d_ws, size_t ws_size,
                              hipStream_t stream) {
    const float* hidden = (const float*)d_in[0];   // [4096, 2560]
    const float* w_qkv  = (const float*)d_in[1];   // [7680, 2560]
    const float* b_qkv  = (const float*)d_in[2];   // [7680]
    const float* w_o    = (const float*)d_in[3];   // [2560, 2560]
    const float* b_o    = (const float*)d_in[4];   // [2560]
    const float* cosp   = (const float*)d_in[5];   // [4096, 20]
    const float* sinp   = (const float*)d_in[6];   // [4096, 20]
    float* out          = (float*)d_out;           // [4096, 2560]

    char* ws = (char*)d_ws;
    size_t off = 0;
    unsigned short* hid_bf  = (unsigned short*)(ws + off); off += (size_t)T_TOK * HIDDEN * 2;
    unsigned short* wqkv_bf = (unsigned short*)(ws + off); off += (size_t)QKV_W * HIDDEN * 2;
    unsigned short* wo_bf   = (unsigned short*)(ws + off); off += (size_t)HIDDEN * HIDDEN * 2;
    unsigned short* qkv_bf  = (unsigned short*)(ws + off); off += (size_t)T_TOK * QKV_W * 2;
    unsigned short* attn_bf = (unsigned short*)(ws + off); off += (size_t)T_TOK * HIDDEN * 2;
    (void)ws_size; (void)n_in; (void)in_sizes; (void)out_size;

    // 1) conversions
    {
        int n;
        n = T_TOK * HIDDEN;
        cvt_f32_bf16<<<(n + 255) / 256, 256, 0, stream>>>(hidden, hid_bf, n);
        n = QKV_W * HIDDEN;
        cvt_f32_bf16<<<(n + 255) / 256, 256, 0, stream>>>(w_qkv, wqkv_bf, n);
        n = HIDDEN * HIDDEN;
        cvt_f32_bf16<<<(n + 255) / 256, 256, 0, stream>>>(w_o, wo_bf, n);
    }
    // 2) QKV projection (bf16 out): 4096x7680 = 64x30 macro tiles
    {
        dim3 grid(QKV_W / 256, T_TOK / 64);
        gemm_wmma_bf16<false><<<grid, 256, 0, stream>>>(hid_bf, wqkv_bf, b_qkv,
                                                        (void*)qkv_bf,
                                                        T_TOK, QKV_W, HIDDEN);
    }
    // 3) RoPE on Q,K
    {
        int n = T_TOK * HEADS * ROT_HALF;
        rope_kernel<<<(n + 255) / 256, 256, 0, stream>>>(qkv_bf, cosp, sinp);
    }
    // 4) flash attention
    {
        dim3 grid(SEQ / 128, HEADS, BATCH);
        flash_attn<<<grid, 256, 0, stream>>>(qkv_bf, attn_bf);
    }
    // 5) output projection (f32 out + bias)
    {
        dim3 grid(HIDDEN / 256, T_TOK / 64);
        gemm_wmma_bf16<true><<<grid, 256, 0, stream>>>(attn_bf, wo_bf, b_o,
                                                       (void*)out,
                                                       T_TOK, HIDDEN, HIDDEN);
    }
}
